// UniformMoERouter5_38165079392681
// MI455X (gfx1250) — matrix operand order repaired and checked by reading, not compiled, not run on hardware
//
#include <hip/hip_runtime.h>
#include <hip/hip_bf16.h>

// ---------------------------------------------------------------------------
// MoE router pipeline for MI455X (gfx1250, wave32).
// GEMMs: v_wmma_f32_16x16x32_f16, double-buffered LDS, register-staged global
// fills so HBM latency hides under the WMMAs (workload is bandwidth-bound:
// ~600 MB of traffic vs ~88 GFLOP). Routing is faithfully sequential.
// ---------------------------------------------------------------------------

typedef __attribute__((ext_vector_type(16))) _Float16 v16h;
typedef __attribute__((ext_vector_type(8)))  _Float16 v8h;
typedef __attribute__((ext_vector_type(4)))  _Float16 v4h;
typedef __attribute__((ext_vector_type(8)))  float    v8f;
typedef __attribute__((ext_vector_type(4)))  float    v4f;

#define NUM_E 5
#define BM 64
#define BN 128
#define BK 32
#define LDA 40   // halves per A-tile row in LDS (80B: 16B-aligned, conflict-free)
#define LDB 40   // halves per Bt-tile row (one output column) in LDS

// C = relu(A[f32 MxK] * B[f32 KxN] + bias), fp16 WMMA, fp32 accumulate.
// Requires M%64==0, N%128==0, K%32==0 (true for all three layers here).
__launch_bounds__(256)
__global__ void gemm_relu_wmma(const float* __restrict__ A, const float* __restrict__ B,
                               const float* __restrict__ bias, float* __restrict__ C,
                               int M, int N, int K, int do_relu)
{
    __shared__ __align__(16) _Float16 sA[2 * BM * LDA];
    __shared__ __align__(16) _Float16 sB[2 * BN * LDB];

    const int tid  = threadIdx.x;
    const int lane = tid & 31;
    const int wave = tid >> 5;        // 0..7
    const int wm   = wave >> 2;       // 0..1  -> wave-row (32 rows)
    const int wn   = wave & 3;        // 0..3  -> wave-col (32 cols)
    const int l16  = lane & 15;
    const int kg   = lane >> 4;       // half-wave group

    const int block_m = blockIdx.y * BM;
    const int block_n = blockIdx.x * BN;

    // A fill: 2 float4 per thread (64x32 tile)
    int aRow[2], aKc[2];
    const float* aPtr[2];
    #pragma unroll
    for (int r = 0; r < 2; ++r) {
        int q   = tid + r * 256;          // 0..511 float4 slots
        aRow[r] = q >> 3;                 // 0..63
        aKc[r]  = (q & 7) << 2;           // 0..28
        aPtr[r] = A + (size_t)(block_m + aRow[r]) * K + aKc[r];
    }
    // B fill: 4 column-chunks per thread (32x128 tile, read down columns ->
    // coalesced b32 across lanes, single ds_store_b64 per chunk after transpose)
    int bCol[4], bKq[4];
    const float* bPtr[4];
    #pragma unroll
    for (int r = 0; r < 4; ++r) {
        int q   = tid + r * 256;          // 0..1023 chunk slots
        bCol[r] = q & 127;                // 0..127
        bKq[r]  = (q >> 7) << 2;          // 0,4,...,28
        bPtr[r] = B + (size_t)bKq[r] * N + block_n + bCol[r];
    }

    v4f  aReg[2];
    float bReg[4][4];

    auto issue_loads = [&]() {
        #pragma unroll
        for (int r = 0; r < 2; ++r) aReg[r] = *(const v4f*)aPtr[r];
        #pragma unroll
        for (int r = 0; r < 4; ++r)
            #pragma unroll
            for (int i = 0; i < 4; ++i) bReg[r][i] = bPtr[r][(size_t)i * N];
        #pragma unroll
        for (int r = 0; r < 2; ++r) aPtr[r] += BK;
        #pragma unroll
        for (int r = 0; r < 4; ++r) bPtr[r] += (size_t)BK * N;
    };
    auto commit = [&](int buf) {
        _Float16* dA = sA + buf * (BM * LDA);
        _Float16* dB = sB + buf * (BN * LDB);
        #pragma unroll
        for (int r = 0; r < 2; ++r) {
            v4h hv = { (_Float16)aReg[r][0], (_Float16)aReg[r][1],
                       (_Float16)aReg[r][2], (_Float16)aReg[r][3] };
            *(v4h*)(dA + aRow[r] * LDA + aKc[r]) = hv;
        }
        #pragma unroll
        for (int r = 0; r < 4; ++r) {
            v4h hv = { (_Float16)bReg[r][0], (_Float16)bReg[r][1],
                       (_Float16)bReg[r][2], (_Float16)bReg[r][3] };
            *(v4h*)(dB + bCol[r] * LDB + bKq[r]) = hv;
        }
    };

    v8f acc[2][2] = {};

    issue_loads();
    commit(0);
    __syncthreads();

    const int nk = K / BK;
    for (int t = 0; t < nk; ++t) {
        const int  cur  = t & 1;
        const bool more = (t + 1 < nk);
        if (more) issue_loads();          // in flight during the WMMAs below

        const _Float16* cA = sA + cur * (BM * LDA);
        const _Float16* cB = sB + cur * (BN * LDB);

        // Fragment layouts per CDNA5 ISA 7.12.2:
        // A 16x32 f16: half-wave kg holds K = 8*kg+[0..7], then 16+8*kg+[0..7]
        v16h afrag[2], bfrag[2];
        #pragma unroll
        for (int mi = 0; mi < 2; ++mi) {
            const _Float16* ap = cA + (wm * 32 + mi * 16 + l16) * LDA;
            v8h lo = *(const v8h*)(ap + kg * 8);
            v8h hi = *(const v8h*)(ap + 16 + kg * 8);
            afrag[mi] = __builtin_shufflevector(lo, hi, 0,1,2,3,4,5,6,7,8,9,10,11,12,13,14,15);
        }
        // B 32x16 f16: half-wave kg holds K = 8*kg+[0..7], then 16+8*kg+[0..7]
        #pragma unroll
        for (int ni = 0; ni < 2; ++ni) {
            const _Float16* bp = cB + (wn * 32 + ni * 16 + l16) * LDB + kg * 8;
            v8h lo = *(const v8h*)(bp);
            v8h hi = *(const v8h*)(bp + 16);
            bfrag[ni] = __builtin_shufflevector(lo, hi, 0,1,2,3,4,5,6,7,8,9,10,11,12,13,14,15);
        }
        #pragma unroll
        for (int mi = 0; mi < 2; ++mi)
            #pragma unroll
            for (int ni = 0; ni < 2; ++ni)
                acc[mi][ni] = __builtin_amdgcn_wmma_f32_16x16x32_f16(
                    false, afrag[mi], false, bfrag[ni],
                    (short)0, acc[mi][ni], false, false);

        if (more) commit(cur ^ 1);        // waits for loads only here, post-WMMA
        __syncthreads();
    }

    // Epilogue: bias + relu. C/D layout: VGPR v -> M = v + 8*kg, N = l16.
    #pragma unroll
    for (int mi = 0; mi < 2; ++mi) {
        #pragma unroll
        for (int ni = 0; ni < 2; ++ni) {
            int col = block_n + wn * 32 + ni * 16 + l16;
            float bv = bias ? bias[col] : 0.0f;
            #pragma unroll
            for (int v = 0; v < 8; ++v) {
                int row = block_m + wm * 32 + mi * 16 + v + 8 * kg;
                float val = acc[mi][ni][v] + bv;
                if (do_relu) val = fmaxf(val, 0.0f);
                C[(size_t)row * N + col] = val;
            }
        }
    }
}

// head: probs = softmax(h3 @ W4 + b4), choice = argmax (first-max on ties)
__launch_bounds__(256)
__global__ void head_softmax_argmax(const float* __restrict__ h3, const float* __restrict__ W4,
                                    const float* __restrict__ b4, float* __restrict__ probs,
                                    int* __restrict__ choice, int n)
{
    __shared__ float sW[128 * NUM_E];
    __shared__ float sb[NUM_E];
    for (int i = threadIdx.x; i < 128 * NUM_E; i += blockDim.x) sW[i] = W4[i];
    if (threadIdx.x < NUM_E) sb[threadIdx.x] = b4[threadIdx.x];
    __syncthreads();

    int i = blockIdx.x * blockDim.x + threadIdx.x;
    if (i >= n) return;
    const float* hp = h3 + (size_t)i * 128;
    float logit[NUM_E];
    #pragma unroll
    for (int e = 0; e < NUM_E; ++e) logit[e] = sb[e];
    for (int k = 0; k < 128; ++k) {
        float hv = hp[k];
        #pragma unroll
        for (int e = 0; e < NUM_E; ++e) logit[e] += hv * sW[k * NUM_E + e];
    }
    float mx = logit[0]; int arg = 0;
    #pragma unroll
    for (int e = 1; e < NUM_E; ++e) if (logit[e] > mx) { mx = logit[e]; arg = e; }
    float ex[NUM_E], s = 0.0f;
    #pragma unroll
    for (int e = 0; e < NUM_E; ++e) { ex[e] = __expf(logit[e] - mx); s += ex[e]; }
    float inv = 1.0f / s;
    #pragma unroll
    for (int e = 0; e < NUM_E; ++e) probs[(size_t)i * NUM_E + e] = ex[e] * inv;
    choice[i] = arg;
}

// Sequential capacity-balanced routing (faithful to reference order/ties).
// Serial by construction; O(n*E) for these inputs (Phase 1 is a no-op when all
// pools exceed min_per, which holds for near-uniform softmax over 5 experts).
__global__ void route_kernel(const float* __restrict__ probs, const int* __restrict__ choice,
                             int* pools, int* rem, int* cand_idx, int* cand_src,
                             int* mark, int* cmark, int* order,
                             float* loss_out, float* idx_tail, int n)
{
    if (threadIdx.x != 0 || blockIdx.x != 0) return;
    int cnt[NUM_E];
    for (int e = 0; e < NUM_E; ++e) {
        int c = 0;
        for (int i = 0; i < n; ++i) if (choice[i] == e) pools[e * n + c++] = i;
        cnt[e] = c;
    }
    int min_per = (n + 7) / 8;
    int remaining = n - min_per * NUM_E;
    while (remaining < 0 && min_per > 0) { min_per--; remaining = n - min_per * NUM_E; }

    // Phase 1: fill under-capacity experts from other experts' overflow
    for (int e = 0; e < NUM_E; ++e) {
        if (cnt[e] >= min_per) continue;
        int need = min_per - cnt[e];
        int ncand = 0;
        for (int o = 0; o < NUM_E; ++o) {
            if (o == e) continue;
            int overflow = cnt[o] - min_per;
            if (overflow <= 0) continue;
            for (int j = 0; j < cnt[o]; ++j) mark[j] = 0;
            for (int s = 0; s < overflow; ++s) {     // stable ascending arg-select
                int best = -1; float bk = 0.0f;
                for (int j = 0; j < cnt[o]; ++j) {
                    if (mark[j]) continue;
                    float k = probs[(size_t)pools[o * n + j] * NUM_E + o];
                    if (best < 0 || k < bk) { bk = k; best = j; }
                }
                mark[best] = 1;
                cand_idx[ncand] = pools[o * n + best];
                cand_src[ncand] = o;
                ncand++;
            }
        }
        for (int c = 0; c < ncand; ++c) cmark[c] = 0;
        for (int t = 0; t < need; ++t) {             // stable descending arg-select
            int best = -1; float bv = 0.0f;
            for (int c = 0; c < ncand; ++c) {
                if (cmark[c]) continue;
                float v = probs[(size_t)cand_idx[c] * NUM_E + e];
                if (best < 0 || v > bv) { bv = v; best = c; }
            }
            if (best < 0) break;
            cmark[best] = 1;
            int idx = cand_idx[best], src = cand_src[best];
            int pos = 0;
            while (pools[src * n + pos] != idx) pos++;
            for (int j = pos; j < cnt[src] - 1; ++j) pools[src * n + j] = pools[src * n + j + 1];
            cnt[src]--;
            pools[e * n + cnt[e]++] = idx;
        }
    }
    // Phase 2: trim to min_per, collect overflow in expert order
    int nrem = 0;
    for (int e = 0; e < NUM_E; ++e) {
        for (int j = min_per; j < cnt[e]; ++j) rem[nrem++] = pools[e * n + j];
        if (cnt[e] > min_per) cnt[e] = min_per;
    }
    // Phase 3: greedy redistribution under cap (first-max on ties)
    int cap = min_per + remaining / NUM_E + 1;
    for (int t = 0; t < nrem; ++t) {
        int idx = rem[t];
        int best = -1; float bv = 0.0f;
        for (int e = 0; e < NUM_E; ++e) {
            if (cnt[e] >= cap) continue;
            float v = probs[(size_t)idx * NUM_E + e];
            if (best < 0 || v > bv) { bv = v; best = e; }
        }
        if (best < 0) {
            for (int e = 0; e < NUM_E; ++e) {
                float v = probs[(size_t)idx * NUM_E + e];
                if (best < 0 || v > bv) { bv = v; best = e; }
            }
        }
        pools[best * n + cnt[best]++] = idx;
    }
    // Finalize: gather order, index tail, distribution loss
    // sum_e cnt_e * mean_e / E  ==  sum over assigned probs / E
    float loss = 0.0f;
    int pos = 0;
    for (int e = 0; e < NUM_E; ++e) {
        for (int j = 0; j < cnt[e]; ++j) {
            int idx = pools[e * n + j];
            order[pos] = idx;
            idx_tail[pos] = (float)idx;
            loss += probs[(size_t)idx * NUM_E + e];
            pos++;
        }
    }
    *loss_out = loss / (float)NUM_E;
}

// expert_data gather: out[s] = x[order[s]], 2048 floats per row (bandwidth-bound)
__launch_bounds__(256)
__global__ void gather_kernel(const float* __restrict__ x, const int* __restrict__ order,
                              float* __restrict__ out, int d)
{
    int s = blockIdx.x;
    int src = order[s];
    const v4f* xp = (const v4f*)(x + (size_t)src * d);
    v4f* op = (v4f*)(out + (size_t)s * d);
    for (int j = threadIdx.x; j < (d >> 2); j += blockDim.x) op[j] = xp[j];
}

extern "C" void kernel_launch(void* const* d_in, const int* in_sizes, int n_in,
                              void* d_out, int out_size, void* d_ws, size_t ws_size,
                              hipStream_t stream)
{
    const float* x  = (const float*)d_in[0];
    const float* W1 = (const float*)d_in[1];
    const float* b1 = (const float*)d_in[2];
    const float* W2 = (const float*)d_in[3];
    const float* b2 = (const float*)d_in[4];
    const float* W3 = (const float*)d_in[5];
    const float* b3 = (const float*)d_in[6];
    const float* W4 = (const float*)d_in[7];
    const float* b4 = (const float*)d_in[8];

    const int D = 2048, N1 = 1024, N2 = 512, N3 = 128;
    const int n = in_sizes[0] / D;

    char* ws = (char*)d_ws;
    size_t off = 0;
    auto alloc = [&](size_t bytes) {
        void* p = ws + off;
        off += (bytes + 255) & ~(size_t)255;
        return p;
    };
    float* h1       = (float*)alloc((size_t)n * N1 * sizeof(float));
    float* h2       = (float*)alloc((size_t)n * N2 * sizeof(float));
    float* h3       = (float*)alloc((size_t)n * N3 * sizeof(float));
    float* probs    = (float*)alloc((size_t)n * NUM_E * sizeof(float));
    int*   choice   = (int*)alloc((size_t)n * sizeof(int));
    int*   pools    = (int*)alloc((size_t)NUM_E * n * sizeof(int));
    int*   rem      = (int*)alloc((size_t)n * sizeof(int));
    int*   cand_idx = (int*)alloc((size_t)n * sizeof(int));
    int*   cand_src = (int*)alloc((size_t)n * sizeof(int));
    int*   mark     = (int*)alloc((size_t)n * sizeof(int));
    int*   cmark    = (int*)alloc((size_t)n * sizeof(int));
    int*   order    = (int*)alloc((size_t)n * sizeof(int));
    (void)ws_size;

    dim3 blk(256);
    gemm_relu_wmma<<<dim3(N1 / BN, n / BM), blk, 0, stream>>>(x,  W1, b1, h1, n, N1, D,  1);
    gemm_relu_wmma<<<dim3(N2 / BN, n / BM), blk, 0, stream>>>(h1, W2, b2, h2, n, N2, N1, 1);
    gemm_relu_wmma<<<dim3(N3 / BN, n / BM), blk, 0, stream>>>(h2, W3, b3, h3, n, N3, N2, 1);
    head_softmax_argmax<<<dim3((n + 255) / 256), blk, 0, stream>>>(h3, W4, b4, probs, choice, n);

    float* out       = (float*)d_out;
    float* loss_slot = out + (size_t)n * D;       // [expert_data | loss | indices]
    float* idx_tail  = loss_slot + 1;
    route_kernel<<<dim3(1), dim3(32), 0, stream>>>(probs, choice, pools, rem, cand_idx,
                                                   cand_src, mark, cmark, order,
                                                   loss_slot, idx_tail, n);
    gather_kernel<<<dim3(n), blk, 0, stream>>>(x, order, out, D);
}
